// MultiHeadAttentionBlock_40072044871790
// MI455X (gfx1250) — matrix operand order repaired and checked by reading, not compiled, not run on hardware
//
#include <hip/hip_runtime.h>
#include <cstdint>
#include <cstddef>

// Problem constants
#define E_DIM 1024
#define NHEAD 16
#define HDIM  64
#define BATCH 4
#define SEQ   1024
#define MTOT  (BATCH * SEQ)   // 4096 rows in all GEMMs
#define QTILES (SEQ / 16)     // 64 query tiles per (b,h)

typedef __bf16 bf16_t;
typedef __attribute__((ext_vector_type(16))) __bf16 v16bf;
typedef __attribute__((ext_vector_type(8)))  __bf16 v8bf;
typedef __attribute__((ext_vector_type(8)))  float  v8f;

// ---- bf16 helpers (RNE) -------------------------------------------------
__device__ __forceinline__ unsigned short f2bf_bits(float f) {
  unsigned u = __builtin_bit_cast(unsigned, f);
  unsigned r = u + 0x7FFFu + ((u >> 16) & 1u);
  return (unsigned short)(r >> 16);
}

// ---- f32 -> bf16 elementwise conversion (for x) -------------------------
__global__ __launch_bounds__(256)
void cvt_f32_bf16(const float* __restrict__ src, unsigned short* __restrict__ dst, int n) {
  int i = blockIdx.x * blockDim.x + threadIdx.x;
  if (i < n) dst[i] = f2bf_bits(src[i]);
}

// ---- f32 -> bf16 64x64 tiled transpose (weights: W[k][n] -> Wt[n][k]) ---
__global__ __launch_bounds__(256)
void transpose_f32_bf16(const float* __restrict__ src, unsigned short* __restrict__ dst,
                        int R, int C) {  // src is R x C, dst is C x R
  __shared__ unsigned short tile[64][65];
  const int c0 = blockIdx.x * 64;
  const int r0 = blockIdx.y * 64;
  const int col = threadIdx.x & 63;
  const int rb  = threadIdx.x >> 6;      // 0..3
#pragma unroll
  for (int i = 0; i < 16; ++i) {
    const int row = rb + i * 4;
    tile[row][col] = f2bf_bits(src[(size_t)(r0 + row) * C + c0 + col]);
  }
  __syncthreads();
#pragma unroll
  for (int i = 0; i < 16; ++i) {
    const int row = rb + i * 4;          // output row within c-tile
    dst[(size_t)(c0 + row) * R + r0 + col] = tile[col][row];
  }
}

// ---- bf16 V[(b,s),(h,d)] -> Vt[b,h,d,s] tiled transpose -----------------
__global__ __launch_bounds__(256)
void transpose_v_bf16(const unsigned short* __restrict__ V, unsigned short* __restrict__ Vt) {
  __shared__ unsigned short tile[64][65];
  const int s0 = blockIdx.x * 64;        // 16 tiles over SEQ
  const int bh = blockIdx.y;             // 0..63
  const int b = bh >> 4, h = bh & 15;
  const int col = threadIdx.x & 63;      // d on load, s on store
  const int rb  = threadIdx.x >> 6;
#pragma unroll
  for (int i = 0; i < 16; ++i) {
    const int row = rb + i * 4;          // s within tile
    tile[row][col] = V[(size_t)(b * SEQ + s0 + row) * E_DIM + h * HDIM + col];
  }
  __syncthreads();
#pragma unroll
  for (int i = 0; i < 16; ++i) {
    const int row = rb + i * 4;          // d
    Vt[((size_t)bh * HDIM + row) * SEQ + s0 + col] = tile[col][row];
  }
}

// ---- GEMM: C[M,N] = A[M,K](bf16) * Bt[N,K](bf16, pre-transposed) + bias -
// Wave tile: 64(M) x 64(N), K-step 32 -> 16 v_wmma_f32_16x16x32_bf16 per
// 8 fragment loads (32 FLOP/byte from L2). Workgroup: 8 waves as 4(M)x2(N)
// -> 256 x 128 block tile. All fragment loads are contiguous 32B/lane.
template <bool STORE_BF16>
__global__ __launch_bounds__(256)
void gemm_bf16_wmma(const unsigned short* __restrict__ A,
                    const unsigned short* __restrict__ Bt,
                    const float* __restrict__ bias,
                    void* __restrict__ Cout,
                    int M, int N, int K) {
  const int lane  = threadIdx.x & 31;
  const int wave  = threadIdx.x >> 5;
  const int waveM = wave >> 1;           // 0..3
  const int waveN = wave & 1;            // 0..1
  const int m0 = blockIdx.x * 256 + waveM * 64;
  const int n0 = blockIdx.y * 128 + waveN * 64;
  const int lm = lane & 15;
  const int kh = lane >> 4;

  v8f acc[4][4];
#pragma unroll
  for (int t = 0; t < 4; ++t)
#pragma unroll
    for (int v = 0; v < 4; ++v)
      acc[t][v] = (v8f){0.f, 0.f, 0.f, 0.f, 0.f, 0.f, 0.f, 0.f};

  const unsigned short* arow = A + (size_t)(m0 + lm) * K + 16 * kh;
  const unsigned short* b0   = Bt + (size_t)(n0 + lm) * K + 16 * kh;

  for (int k0 = 0; k0 < K; k0 += 32) {
    v16bf af[4];
#pragma unroll
    for (int t = 0; t < 4; ++t) {
      const unsigned short* ap = arow + (size_t)(16 * t) * K + k0;
      v8bf lo = *(const v8bf*)(ap);
      v8bf hi = *(const v8bf*)(ap + 8);
#pragma unroll
      for (int e = 0; e < 8; ++e) { af[t][e] = lo[e]; af[t][8 + e] = hi[e]; }
    }
    v16bf bfr[4];
#pragma unroll
    for (int v = 0; v < 4; ++v)
      bfr[v] = *(const v16bf*)(b0 + (size_t)(16 * v) * K + k0);

#pragma unroll
    for (int t = 0; t < 4; ++t)
#pragma unroll
      for (int v = 0; v < 4; ++v)
        acc[t][v] = __builtin_amdgcn_wmma_f32_16x16x32_bf16(
            false, af[t], false, bfr[v], (short)0, acc[t][v], false, false);
  }

  // Epilogue: bias + store (C-layout: row = r + 8*kh, col = lm)
#pragma unroll
  for (int v = 0; v < 4; ++v) {
    const int n = n0 + 16 * v + lm;
    const float bv = bias[n];
#pragma unroll
    for (int t = 0; t < 4; ++t) {
#pragma unroll
      for (int r = 0; r < 8; ++r) {
        const int m = m0 + 16 * t + r + 8 * kh;
        const float val = acc[t][v][r] + bv;
        if (STORE_BF16)
          ((unsigned short*)Cout)[(size_t)m * N + n] = f2bf_bits(val);
        else
          ((float*)Cout)[(size_t)m * N + n] = val;
      }
    }
  }
}

// ---- Causal flash attention ---------------------------------------------
// One wave handles 16 queries x full head-dim (64) for one (b,h).
// Online softmax in base-2 domain; P converted C-layout -> A-layout via LDS.
// V is pre-transposed so P*V B-fragments are contiguous vector loads.
__global__ __launch_bounds__(128)
void attention_wmma(const unsigned short* __restrict__ Q,
                    const unsigned short* __restrict__ Kmat,
                    const unsigned short* __restrict__ Vt,
                    unsigned short* __restrict__ O) {
  __shared__ unsigned short pbuf[4][16 * 40];   // 40-half row pitch (bank pad)

  const int lane = threadIdx.x & 31;
  const int wave = threadIdx.x >> 5;
  const int w  = blockIdx.x * 4 + wave;         // flat wave id, 0..4095
  const int qt = w & (QTILES - 1);
  const int h  = (w >> 6) & (NHEAD - 1);
  const int b  = w >> 10;
  const int q0 = qt * 16;
  const int lm = lane & 15;
  const int kh = lane >> 4;
  // fold 1/sqrt(D) and log2(e) into the scores (softmax base-2 domain)
  const float kscale = 0.125f * 1.44269504088896340736f;

  // Q fragments (A-layout), d-chunks {0..31} and {32..63}; kept in VGPRs
  v16bf qf[2];
#pragma unroll
  for (int t = 0; t < 2; ++t) {
    const unsigned short* qp =
        Q + (size_t)(b * SEQ + q0 + lm) * E_DIM + h * HDIM + 32 * t + 16 * kh;
    v8bf lo = *(const v8bf*)(qp);
    v8bf hi = *(const v8bf*)(qp + 8);
#pragma unroll
    for (int e = 0; e < 8; ++e) { qf[t][e] = lo[e]; qf[t][8 + e] = hi[e]; }
  }

  v8f acc[4];
#pragma unroll
  for (int v = 0; v < 4; ++v)
    acc[v] = (v8f){0.f, 0.f, 0.f, 0.f, 0.f, 0.f, 0.f, 0.f};
  float mrow[8], lrow[8];
#pragma unroll
  for (int r = 0; r < 8; ++r) { mrow[r] = -1e30f; lrow[r] = 0.f; }

  unsigned short* P = pbuf[wave];
  const unsigned short* kbase =
      Kmat + (size_t)(b * SEQ) * E_DIM + h * HDIM + 16 * kh;
  const unsigned short* vbase =
      Vt + ((size_t)((b * NHEAD + h) * HDIM) + lm) * SEQ + 16 * kh;
  const int qend = q0 + 15;

  for (int j0 = 0; j0 <= qend; j0 += 32) {
    // ---- scores: S = Q * K^T for keys [j0, j0+32) ----
    v8f s[2];
#pragma unroll
    for (int t = 0; t < 2; ++t) {
      const unsigned short* kp = kbase + (size_t)(j0 + 16 * t + lm) * E_DIM;
      v16bf kf0 = *(const v16bf*)(kp);        // d = 0..31 slice
      v16bf kf1 = *(const v16bf*)(kp + 32);   // d = 32..63 slice
      v8f z = (v8f){0.f, 0.f, 0.f, 0.f, 0.f, 0.f, 0.f, 0.f};
      z = __builtin_amdgcn_wmma_f32_16x16x32_bf16(false, qf[0], false, kf0,
                                                  (short)0, z, false, false);
      z = __builtin_amdgcn_wmma_f32_16x16x32_bf16(false, qf[1], false, kf1,
                                                  (short)0, z, false, false);
      s[t] = z;
    }

    // ---- causal mask + online softmax (row = r + 8*kh, col = lm) ----
#pragma unroll
    for (int r = 0; r < 8; ++r) {
      const int qrow = q0 + r + 8 * kh;
      float s0 = s[0][r] * kscale;
      float s1 = s[1][r] * kscale;
      if (j0 + lm > qrow)      s0 = -1e30f;
      if (j0 + 16 + lm > qrow) s1 = -1e30f;
      float loc = fmaxf(s0, s1);
      loc = fmaxf(loc, __shfl_xor(loc, 1));
      loc = fmaxf(loc, __shfl_xor(loc, 2));
      loc = fmaxf(loc, __shfl_xor(loc, 4));
      loc = fmaxf(loc, __shfl_xor(loc, 8));
      const float mnew = fmaxf(mrow[r], loc);
      const float cf = exp2f(mrow[r] - mnew);
      const float p0 = exp2f(s0 - mnew);
      const float p1 = exp2f(s1 - mnew);
      float ps = p0 + p1;
      ps += __shfl_xor(ps, 1);
      ps += __shfl_xor(ps, 2);
      ps += __shfl_xor(ps, 4);
      ps += __shfl_xor(ps, 8);
      lrow[r] = lrow[r] * cf + ps;
      mrow[r] = mnew;
#pragma unroll
      for (int v = 0; v < 4; ++v) acc[v][r] *= cf;
      // stash P tile (16x32) to LDS in logical (row, key) order
      P[(r + 8 * kh) * 40 + lm]      = f2bf_bits(p0);
      P[(r + 8 * kh) * 40 + 16 + lm] = f2bf_bits(p1);
    }

    // ---- P (A-layout from LDS) x V (B-layout, contiguous from Vt) ----
    v16bf pf;
    {
      const unsigned short* pp = P + lm * 40;
#pragma unroll
      for (int e = 0; e < 16; ++e) {
        const int kk = e + 16 * kh;
        pf[e] = __builtin_bit_cast(bf16_t, pp[kk]);
      }
    }
#pragma unroll
    for (int v = 0; v < 4; ++v) {
      v16bf vf = *(const v16bf*)(vbase + (size_t)(16 * v) * SEQ + j0);
      acc[v] = __builtin_amdgcn_wmma_f32_16x16x32_bf16(
          false, pf, false, vf, (short)0, acc[v], false, false);
    }
  }

  // ---- normalize and store O[b, s, h*D + d] as bf16 ----
#pragma unroll
  for (int r = 0; r < 8; ++r) {
    const float inv = 1.0f / lrow[r];
    const int m = q0 + r + 8 * kh;
#pragma unroll
    for (int v = 0; v < 4; ++v)
      O[(size_t)(b * SEQ + m) * E_DIM + h * HDIM + 16 * v + lm] =
          f2bf_bits(acc[v][r] * inv);
  }
}

// ---- launcher ------------------------------------------------------------
extern "C" void kernel_launch(void* const* d_in, const int* in_sizes, int n_in,
                              void* d_out, int out_size, void* d_ws, size_t ws_size,
                              hipStream_t stream) {
  (void)in_sizes; (void)n_in; (void)out_size; (void)ws_size;
  const float* x  = (const float*)d_in[0];
  const float* Wq = (const float*)d_in[1];
  const float* bq = (const float*)d_in[2];
  const float* Wk = (const float*)d_in[3];
  const float* bk = (const float*)d_in[4];
  const float* Wv = (const float*)d_in[5];
  const float* bv = (const float*)d_in[6];
  const float* Wo = (const float*)d_in[7];
  const float* bo = (const float*)d_in[8];
  float* out = (float*)d_out;

  // workspace layout (bf16 = ushort)
  char* ws = (char*)d_ws;
  const size_t SZ_X = (size_t)MTOT * E_DIM * 2;     // 8 MB
  const size_t SZ_W = (size_t)E_DIM * E_DIM * 2;    // 2 MB
  unsigned short* xb  = (unsigned short*)(ws);
  unsigned short* Wqt = (unsigned short*)(ws + SZ_X);
  unsigned short* Wkt = (unsigned short*)(ws + SZ_X + 1 * SZ_W);
  unsigned short* Wvt = (unsigned short*)(ws + SZ_X + 2 * SZ_W);
  unsigned short* Wot = (unsigned short*)(ws + SZ_X + 3 * SZ_W);
  unsigned short* Qb  = (unsigned short*)(ws + SZ_X + 4 * SZ_W);
  unsigned short* Kb  = (unsigned short*)(ws + 2 * SZ_X + 4 * SZ_W);
  unsigned short* Vb  = (unsigned short*)(ws + 3 * SZ_X + 4 * SZ_W);
  unsigned short* Vtb = (unsigned short*)(ws + 4 * SZ_X + 4 * SZ_W);
  unsigned short* Ob  = (unsigned short*)(ws + 5 * SZ_X + 4 * SZ_W);

  // 1) precision conversion (+ weight transpose for contiguous B-fragments)
  const int nx = MTOT * E_DIM;
  cvt_f32_bf16<<<(nx + 255) / 256, 256, 0, stream>>>(x, xb, nx);
  dim3 tg(E_DIM / 64, E_DIM / 64);
  transpose_f32_bf16<<<tg, 256, 0, stream>>>(Wq, Wqt, E_DIM, E_DIM);
  transpose_f32_bf16<<<tg, 256, 0, stream>>>(Wk, Wkt, E_DIM, E_DIM);
  transpose_f32_bf16<<<tg, 256, 0, stream>>>(Wv, Wvt, E_DIM, E_DIM);
  transpose_f32_bf16<<<tg, 256, 0, stream>>>(Wo, Wot, E_DIM, E_DIM);

  // 2) QKV projections (bf16 output for attention fragments)
  dim3 gg(MTOT / 256, E_DIM / 128);
  gemm_bf16_wmma<true><<<gg, 256, 0, stream>>>(xb, Wqt, bq, Qb, MTOT, E_DIM, E_DIM);
  gemm_bf16_wmma<true><<<gg, 256, 0, stream>>>(xb, Wkt, bk, Kb, MTOT, E_DIM, E_DIM);
  gemm_bf16_wmma<true><<<gg, 256, 0, stream>>>(xb, Wvt, bv, Vb, MTOT, E_DIM, E_DIM);

  // 2b) V -> Vt[b,h,d,s] so P*V fragments are contiguous
  dim3 tv(SEQ / 64, BATCH * NHEAD);
  transpose_v_bf16<<<tv, 256, 0, stream>>>(Vb, Vtb);

  // 3) causal flash attention (4096 waves, 4 waves/block)
  attention_wmma<<<(BATCH * NHEAD * QTILES) / 4, 128, 0, stream>>>(Qb, Kb, Vtb, Ob);

  // 4) output projection -> f32 result
  gemm_bf16_wmma<false><<<gg, 256, 0, stream>>>(Ob, Wot, bo, out, MTOT, E_DIM, E_DIM);
}
